// BidirLayer_90460601189067
// MI455X (gfx1250) — matrix operand order repaired and checked by reading, not compiled, not run on hardware
//
#include <hip/hip_runtime.h>

typedef __attribute__((ext_vector_type(16))) __bf16 v16bf;
typedef __attribute__((ext_vector_type(8)))  __bf16 v8bf;
typedef __attribute__((ext_vector_type(4)))  __bf16 v4bf;
typedef __attribute__((ext_vector_type(2)))  __bf16 v2bf;
typedef __attribute__((ext_vector_type(8)))  float  v8f;

#define SS 1024
#define BB 64
#define II 512
#define HH 512

// Packed f32 -> bf16 convert: hardware v_cvt_pk_bf16_f32 when available.
static __device__ __forceinline__ v2bf pk2(float lo, float hi) {
#if __has_builtin(__builtin_amdgcn_cvt_pk_bf16_f32)
    return __builtin_amdgcn_cvt_pk_bf16_f32(lo, hi);
#else
    v2bf r; r[0] = (__bf16)lo; r[1] = (__bf16)hi; return r;
#endif
}

// ---------------------------------------------------------------------------
// Kernel 0a: elementwise f32 -> bf16 for the activations (one-time).
// ---------------------------------------------------------------------------
__global__ __launch_bounds__(256) void cvt_kernel(const float* __restrict__ src,
                                                  __bf16* __restrict__ dst, int n4) {
    int i = blockIdx.x * blockDim.x + threadIdx.x;
    const int stride = gridDim.x * blockDim.x;
    for (; i < n4; i += stride) {
        float4 x = reinterpret_cast<const float4*>(src)[i];
        v2bf a = pk2(x.x, x.y);
        v2bf b = pk2(x.z, x.w);
        v4bf v; v[0] = a[0]; v[1] = a[1]; v[2] = b[0]; v[3] = b[1];
        reinterpret_cast<v4bf*>(dst)[i] = v;
    }
}

// ---------------------------------------------------------------------------
// Kernel 0b: pack a [2][512][512] f32 weight matrix into per-lane WMMA-A
// layout: apk[dir][ht][kc][lane][e] (bf16, 32B contiguous per lane per chunk).
// Contiguous k order used here (not AMD's table order): lane<16 holds M=lane%16, K = k0+0..15;
// lane>=16 holds K = k0+16..31.
// ---------------------------------------------------------------------------
__global__ __launch_bounds__(256) void pack_a_kernel(const float* __restrict__ W,
                                                     __bf16* __restrict__ apk,
                                                     int total) {
    int idx = blockIdx.x * blockDim.x + threadIdx.x;
    const int stride = gridDim.x * blockDim.x;
    for (; idx < total; idx += stride) {
        const int e    = idx & 15;
        const int lane = (idx >> 4) & 31;
        const int kc   = (idx >> 9) & 15;
        const int ht   = (idx >> 13) & 31;
        const int dir  = idx >> 18;
        const int half = lane >> 4;
        const int m    = ht * 16 + (lane & 15);
        const int k    = kc * 32 + (half ? 16 : 0) + e;
        apk[idx] = (__bf16)W[((size_t)dir * HH + m) * 512 + k];
    }
}

// ---------------------------------------------------------------------------
// Kernel 1: xwT[dir][s][h][b] = sum_i Wih[dir][h][i]*inp[s][b][i] + bih + bhh
// grid = 2 dirs * 32 h-tiles * 64 s-chunks = 4096 blocks, 256 threads (8 waves).
// Packed A tile staged in LDS once, shared by all 8 waves.
// Two accumulator chains halve the WMMA RAW dependency depth.
// ---------------------------------------------------------------------------
__global__ __launch_bounds__(256) void proj_kernel(
    const __bf16* __restrict__ inp_bf,  // [S][B][I] bf16
    const __bf16* __restrict__ wih_pk,  // [2][32][16][32][16] packed bf16
    const float*  __restrict__ bih,     // [2][H]
    const float*  __restrict__ bhh,     // [2][H]
    float* __restrict__ xwT)            // [2][S][H][B]
{
    __shared__ __bf16 a_lds[16 * 512];  // packed A tile, 16 KB

    const int lane = threadIdx.x & 31;
    const int wave = threadIdx.x >> 5;
    const int half = lane >> 4;
    const int ln   = lane & 15;

    int bid = blockIdx.x;
    const int schunk = bid & 63;  bid >>= 6;
    const int ht     = bid & 31;  bid >>= 5;
    const int dir    = bid;
    const int h0     = ht * 16;

    // cooperative copy of the packed A tile into LDS (16 KB contiguous)
    const __bf16* asrc = wih_pk + (size_t)(dir * 32 + ht) * 8192;
    for (int i = threadIdx.x; i < 1024; i += 256)
        reinterpret_cast<v8bf*>(a_lds)[i] =
            reinterpret_cast<const v8bf*>(asrc)[i];

    float biasr[8];
    #pragma unroll
    for (int r = 0; r < 8; ++r) {
        int m = h0 + r + 8 * half;
        biasr[r] = bih[dir * HH + m] + bhh[dir * HH + m];
    }
    __syncthreads();

    // 16 timesteps x 4 batch-tiles = 64 output tiles per block, 8 per wave.
    for (int t = wave; t < 64; t += 8) {
        const int sl = t >> 2;
        const int bt = t & 3;
        const int s  = schunk * 16 + sl;
        // B operand, same contiguous k order as packed A: lane holds 16 contiguous K for column N=lane%16.
        const __bf16* brow = inp_bf + ((size_t)s * BB + bt * 16 + ln) * II
                                    + (half ? 16 : 0);
        v8f acc0 = {}, acc1 = {};
        #pragma unroll 4
        for (int kc = 0; kc < 8; ++kc) {
            v16bf a0 = *reinterpret_cast<const v16bf*>(&a_lds[((2 * kc)     * 32 + lane) * 16]);
            v16bf b0 = *reinterpret_cast<const v16bf*>(brow + (2 * kc) * 32);
            v16bf a1 = *reinterpret_cast<const v16bf*>(&a_lds[((2 * kc + 1) * 32 + lane) * 16]);
            v16bf b1 = *reinterpret_cast<const v16bf*>(brow + (2 * kc + 1) * 32);
            acc0 = __builtin_amdgcn_wmma_f32_16x16x32_bf16(false, a0, false, b0,
                                                           (short)0, acc0, false, false);
            acc1 = __builtin_amdgcn_wmma_f32_16x16x32_bf16(false, a1, false, b1,
                                                           (short)0, acc1, false, false);
        }
        // C layout: VGPR r -> (m = h0 + r + 8*half, n = lane%16); store transposed.
        float* orow = xwT + ((size_t)(dir * SS + s) * HH) * BB;
        #pragma unroll
        for (int r = 0; r < 8; ++r) {
            int m = h0 + r + 8 * half;
            orow[(size_t)m * BB + bt * 16 + ln] = acc0[r] + acc1[r] + biasr[r];
        }
    }
}

// ---------------------------------------------------------------------------
// Kernel 2: sequential scan. grid = 2 dirs * 4 batch-tiles = 8 blocks,
// 1024 threads = 32 waves; wave w owns h-dims [16w, 16w+16).
// Packed Whh slice streamed from L2 each step (1 MB total, always L2-hot);
// h kept in LDS as bf16 with padded rows (bank-conflict free B reads).
// Two accumulator chains halve the per-step WMMA dependency depth.
// ---------------------------------------------------------------------------
#define HPAD 520   // 1040 B row stride: 16B aligned, 4-bank shift per row

__global__ __launch_bounds__(1024) void scan_kernel(
    const float*  __restrict__ xwT,     // [2][S][H][B]
    const float*  __restrict__ states,  // [2][B][H]
    const __bf16* __restrict__ whh_pk,  // [2][32][16][32][16] packed bf16
    float* __restrict__ out,            // [S][B][2H]
    float* __restrict__ out_states)     // [2][B][H]
{
    __shared__ __bf16 hlds[16][HPAD];   // h state for this batch-tile

    const int wave = threadIdx.x >> 5;  // 0..31
    const int lane = threadIdx.x & 31;
    const int half = lane >> 4;
    const int ln   = lane & 15;
    const int dir  = blockIdx.x >> 2;
    const int b0   = (blockIdx.x & 3) * 16;
    const int h0   = wave * 16;

    // this wave's packed Whh slice (per-lane 32B runs, contiguous & L2-resident)
    const __bf16* wpk = whh_pk + (size_t)(dir * 32 + wave) * 8192 + lane * 16;

    // init h in LDS from initial states
    for (int idx = threadIdx.x; idx < 16 * HH; idx += 1024) {
        int b = idx >> 9, k = idx & (HH - 1);
        hlds[b][k] = (__bf16)states[((size_t)dir * BB + b0 + b) * HH + k];
    }
    __syncthreads();

    for (int step = 0; step < SS; ++step) {
        const int s = dir ? (SS - 1 - step) : step;
        const float* xrow = xwT + ((size_t)(dir * SS + s) * HH) * BB + b0;

        if (step + 1 < SS) {  // prefetch next timestep's xw tile (global_prefetch_b8)
            const int s2 = dir ? (SS - 2 - step) : (step + 1);
            __builtin_prefetch(xwT + ((size_t)(dir * SS + s2) * HH + h0) * BB + b0, 0, 1);
        }

        v8f acc0 = {}, acc1 = {};
        const __bf16* hb = &hlds[ln][half ? 16 : 0];
        #pragma unroll 4
        for (int kc = 0; kc < 8; ++kc) {
            v16bf a0 = *reinterpret_cast<const v16bf*>(wpk + (2 * kc)     * 512);
            v16bf b0 = *reinterpret_cast<const v16bf*>(hb  + (2 * kc)     * 32);
            v16bf a1 = *reinterpret_cast<const v16bf*>(wpk + (2 * kc + 1) * 512);
            v16bf b1 = *reinterpret_cast<const v16bf*>(hb  + (2 * kc + 1) * 32);
            acc0 = __builtin_amdgcn_wmma_f32_16x16x32_bf16(false, a0, false, b0,
                                                           (short)0, acc0, false, false);
            acc1 = __builtin_amdgcn_wmma_f32_16x16x32_bf16(false, a1, false, b1,
                                                           (short)0, acc1, false, false);
        }

        float hv[8];
        #pragma unroll
        for (int r = 0; r < 8; ++r) {
            const int m = h0 + r + 8 * half;
            hv[r] = tanhf(acc0[r] + acc1[r] + xrow[(size_t)m * BB + ln]);
            out[((size_t)s * BB + b0 + ln) * (2 * HH) + dir * HH + m] = hv[r];
        }

        __syncthreads();   // everyone done reading old h
        // lane writes its 8 new h values: one contiguous 16-byte bf16 run
        v2bf p0 = pk2(hv[0], hv[1]), p1 = pk2(hv[2], hv[3]);
        v2bf p2 = pk2(hv[4], hv[5]), p3 = pk2(hv[6], hv[7]);
        v8bf pk; pk[0]=p0[0]; pk[1]=p0[1]; pk[2]=p1[0]; pk[3]=p1[1];
                 pk[4]=p2[0]; pk[5]=p2[1]; pk[6]=p3[0]; pk[7]=p3[1];
        *reinterpret_cast<v8bf*>(&hlds[ln][h0 + 8 * half]) = pk;
        __syncthreads();   // new h visible before next step
    }

    // final states
    for (int idx = threadIdx.x; idx < 16 * HH; idx += 1024) {
        int b = idx >> 9, k = idx & (HH - 1);
        out_states[((size_t)dir * BB + b0 + b) * HH + k] = (float)hlds[b][k];
    }
}

extern "C" void kernel_launch(void* const* d_in, const int* in_sizes, int n_in,
                              void* d_out, int out_size, void* d_ws, size_t ws_size,
                              hipStream_t stream) {
    const float* inp    = (const float*)d_in[0];   // [S][B][I]
    const float* states = (const float*)d_in[1];   // [2][B][H]
    const float* Wih    = (const float*)d_in[2];   // [2][H][I]
    const float* Whh    = (const float*)d_in[3];   // [2][H][H]
    const float* bih    = (const float*)d_in[4];   // [2][H]
    const float* bhh    = (const float*)d_in[5];   // [2][H]

    float* out        = (float*)d_out;                       // [S][B][2H]
    float* out_states = out + (size_t)SS * BB * 2 * HH;      // [2][B][H]

    // workspace layout
    char* ws = (char*)d_ws;
    float*  xwT    = (float*)ws;                              // 2*S*H*B*4 = 256 MB
    ws += (size_t)2 * SS * HH * BB * sizeof(float);
    __bf16* inp_bf = (__bf16*)ws;                             // S*B*I*2   =  64 MB
    ws += (size_t)SS * BB * II * sizeof(__bf16);
    __bf16* wih_pk = (__bf16*)ws;                             // 1 MB packed
    ws += (size_t)2 * HH * II * sizeof(__bf16);
    __bf16* whh_pk = (__bf16*)ws;                             // 1 MB packed

    const int wtotal = 2 * 32 * 16 * 32 * 16;                 // 524288 per matrix

    cvt_kernel<<<8192, 256, 0, stream>>>(inp, inp_bf, (SS * BB * II) / 4);
    pack_a_kernel<<<2048, 256, 0, stream>>>(Wih, wih_pk, wtotal);
    pack_a_kernel<<<2048, 256, 0, stream>>>(Whh, whh_pk, wtotal);

    proj_kernel<<<4096, 256, 0, stream>>>(inp_bf, wih_pk, bih, bhh, xwT);
    scan_kernel<<<8, 1024, 0, stream>>>(xwT, states, whh_pk, out, out_states);
}
